// MaskEstimation_24807731102197
// MI455X (gfx1250) — matrix operand order repaired and checked by reading, not compiled, not run on hardware
//
#include <hip/hip_runtime.h>

#define NB 30
#define NN 128      // feature dim
#define HH 256      // hidden dim (2N)
#define NBATCH 8
#define TTOT 2000
#define TCH 80      // timesteps per workgroup (2000 = 25 * 80)
#define NCHUNK 25
#define MT 5        // 16-row M tiles per chunk
#define OUTC 1900
#define LN_EPS 1e-5f

// ---- LDS layout (bytes) ----
//  [0     .. 40960)  region A: first float xs[128][80] (40960 B, TDM-written, contiguous rows)
//                              then  hsF: 5*8 A-fragments * 1024 B = 40960 B (GEMM2 A frags)
//  [40960 .. 61440)  ysF: 5*4 A-fragments * 1024 B = 20480 B (GEMM1 A frags)
#define SMEM_BYTES 61440
#define XS_STRIDE 80
#define YSF_OFF 40960

typedef __attribute__((ext_vector_type(16))) __bf16        v16bf;
typedef __attribute__((ext_vector_type(8)))  float         v8f;
typedef __attribute__((ext_vector_type(8)))  unsigned int  v8u;
typedef __attribute__((ext_vector_type(4)))  unsigned int  u32x4;
typedef __attribute__((ext_vector_type(8)))  int           i32x8;
typedef __attribute__((ext_vector_type(4)))  int           i32x4;

union Frag16 {
    v16bf v;
    v8u   u8;
};

struct BandArgs {
    const float* w2[NB];
    const float* b2[NB];
    int off[NB];    // output column offset per band
    int n2[NB];     // 2*bandwidth per band
    int w2fo[NB];   // dword offset of band's packed W2 fragments in ws
};

__device__ __forceinline__ unsigned short f2bf(float f) {
    // native f32 -> bf16 (RNE) via hardware cvt
    union { __bf16 b; unsigned short s; } u;
    u.b = (__bf16)f;
    return u.s;
}

// ---------------------------------------------------------------------------
// Pre-pack W1 (f32 [NB,128,256]) into WMMA B-fragment order (bf16) in ws.
// Per band: 16 n-tiles * 4 k-blocks = 64 frags * 256 dwords = 16384 dwords.
// B-fragment layout (16-bit, 32x16): lane = n + 16*((k>>3)&1), vgpr j = 4*((k>>4)&1) + ((k&7)>>1).
// ---------------------------------------------------------------------------
__global__ __launch_bounds__(256) void prepack_w1(const float* __restrict__ w1,
                                                  unsigned int* __restrict__ wsW1) {
    int d = blockIdx.x * 256 + threadIdx.x;     // < 30*16384 = 491520
    int band = d >> 14;
    int r = d & 16383;
    int frag = r >> 8;                          // 0..63
    int lane = (r >> 3) & 31;
    int j = r & 7;
    int nt = frag >> 2, kb = frag & 3;
    int n = nt * 16 + (lane & 15);
    int k = kb * 32 + ((lane >> 4) << 3) + ((j >> 2) << 4) + (j & 3) * 2;
    const float* wb = w1 + ((size_t)band * NN + k) * HH + n;
    unsigned int lo = f2bf(wb[0]);
    unsigned int hi = f2bf(wb[HH]);
    wsW1[d] = lo | (hi << 16);
}

// ---------------------------------------------------------------------------
// Pre-pack ragged W2 (f32 [256, N2] per band) into B-fragment order, zero-pad
// columns >= N2. Per band: n2t n-tiles * 8 k-blocks frags.
// ---------------------------------------------------------------------------
__global__ __launch_bounds__(256) void prepack_w2(BandArgs ba,
                                                  unsigned int* __restrict__ wsW2) {
    int band = blockIdx.y;
    int d = blockIdx.x * 256 + threadIdx.x;
    int N2 = ba.n2[band];
    int n2t = (N2 + 15) >> 4;
    if (d >= n2t * 8 * 256) return;
    int frag = d >> 8;
    int lane = (d >> 3) & 31;
    int j = d & 7;
    int nt = frag >> 3, kb = frag & 7;
    int n = nt * 16 + (lane & 15);
    int k = kb * 32 + ((lane >> 4) << 3) + ((j >> 2) << 4) + (j & 3) * 2;
    const float* w2 = ba.w2[band];
    unsigned int lo = 0u, hi = 0u;
    if (n < N2) {
        lo = f2bf(w2[(size_t)k * N2 + n]);
        hi = f2bf(w2[(size_t)(k + 1) * N2 + n]);
    }
    wsW2[(size_t)ba.w2fo[band] + d] = lo | (hi << 16);
}

// ---------------------------------------------------------------------------
// Fused main kernel: TDM stage -> LN -> GEMM1(+ReLU) -> ragged GEMM2.
// ---------------------------------------------------------------------------
__global__ __launch_bounds__(256) void mask_estimation_fused(
    const float* __restrict__ x,       // [B, NB, N, T]
    const float* __restrict__ gamma,   // [NB, N]
    const float* __restrict__ beta,    // [NB, N]
    const float* __restrict__ b1,      // [NB, 2N]
    const unsigned int* __restrict__ wsW1,  // packed W1 B-frags
    const unsigned int* __restrict__ wsW2,  // packed W2 B-frags
    float* __restrict__ out,           // [B, T, 1900] flat
    BandArgs ba)
{
    __shared__ __align__(16) unsigned char smem[SMEM_BYTES];
    float*          xs    = (float*)smem;                     // [128][80]
    unsigned int*   hsF   = (unsigned int*)smem;              // 40 frags * 256 dw
    unsigned short* hsF16 = (unsigned short*)smem;
    unsigned int*   ysF   = (unsigned int*)(smem + YSF_OFF);  // 20 frags * 256 dw
    unsigned short* ysF16 = (unsigned short*)(smem + YSF_OFF);

    const int bx   = blockIdx.x;
    const int band = bx / (NBATCH * NCHUNK);
    const int rem  = bx % (NBATCH * NCHUNK);
    const int bat  = rem / NCHUNK;
    const int t0   = (rem % NCHUNK) * TCH;

    const int tid  = threadIdx.x;
    const int lane = tid & 31;
    const int wave = tid >> 5;
    const int lc   = lane & 15;
    const int lhi  = (lane >> 4) & 1;

    // ---------- Stage 1: TDM async stage of x tile [128 n][80 t] into LDS ----------
    // 2D tile: dim0 (contiguous) = t (80 dwords), dim1 = n (128 rows, stride 2000 dwords).
    if (wave == 0) {
        const float* xg = x + ((size_t)(bat * NB + band) * NN) * TTOT + t0;
        unsigned long long ga = (unsigned long long)(uintptr_t)xg;   // global byte address
        unsigned int lds_off = (unsigned int)(uintptr_t)(void*)smem; // low 32b of LDS generic addr = LDS offset

        u32x4 g0;
        g0[0] = 1u;                                            // count=1 (valid descriptor)
        g0[1] = lds_off;                                       // lds_addr (bytes)
        g0[2] = (unsigned int)(ga & 0xFFFFFFFFull);            // global_addr[31:0]
        g0[3] = (unsigned int)((ga >> 32) & 0x01FFFFFFull)     // global_addr[56:32]
                | 0x80000000u;                                 // type = 2 ("image")

        i32x8 g1;
        g1[0] = 0x00020000;                  // data_size = 2 (4 bytes); no mask/pad/iterate
        g1[1] = (int)((unsigned)TTOT << 16); // tensor_dim0[15:0] in [31:16]
        g1[2] = (int)((unsigned)NN << 16);   // tensor_dim0[31:16]=0 | tensor_dim1[15:0]
        g1[3] = (int)((unsigned)TCH << 16);  // tensor_dim1[31:16]=0 | tile_dim0=80
        g1[4] = NN;                          // tile_dim1=128 | tile_dim2=0
        g1[5] = TTOT;                        // tensor_dim0_stride[31:0] = 2000
        g1[6] = 0;                           // stride hi | dim1_stride lo
        g1[7] = 0;

        i32x4 z4 = {0, 0, 0, 0};
        i32x8 z8 = {0, 0, 0, 0, 0, 0, 0, 0};
        __builtin_amdgcn_tensor_load_to_lds(g0, g1, z4, z4, z8, 0);
        __builtin_amdgcn_s_wait_tensorcnt(0);
    }
    __syncthreads();

    // ---------- Stage 2: LayerNorm per t; write bf16 directly in A-frag order ----------
    if (tid < TCH) {
        const int t = tid;
        float s = 0.f, s2 = 0.f;
        for (int n = 0; n < NN; ++n) {
            float v = xs[n * XS_STRIDE + t];
            s += v; s2 += v * v;
        }
        const float mean = s * (1.0f / NN);
        const float var  = s2 * (1.0f / NN) - mean * mean;
        const float inv  = rsqrtf(var + LN_EPS);
        const float* gb = gamma + band * NN;
        const float* bb = beta  + band * NN;
        const int mtBase = (t >> 4) * 4;
        const int mlow   = t & 15;
        for (int n = 0; n < NN; ++n) {
            float v = (xs[n * XS_STRIDE + t] - mean) * inv * gb[n] + bb[n];
            // A-frag layout: lane = m + 16*((k>>3)&1); j = 4*((k>>4)&1) + ((k&7)>>1)
            int fb    = mtBase + (n >> 5);
            int laneA = mlow + (((n >> 3) & 1) << 4);
            int j     = (((n >> 4) & 1) << 2) + ((n & 7) >> 1);
            ysF16[fb * 512 + laneA * 16 + j * 2 + (n & 1)] = f2bf(v);
        }
    }
    __syncthreads();   // xs dead; hsF region reusable

    // ---------- Stage 3: GEMM1  h = relu(y[80x128] @ W1[128x256] + b1) ----------
    const unsigned int* w1f = wsW1 + band * 16384;
    const float* b1b = b1 + band * HH;
    for (int tile = wave; tile < MT * 16; tile += 8) {
        const int mt = tile >> 4;
        const int nt = tile & 15;
        v8f c = {0.f, 0.f, 0.f, 0.f, 0.f, 0.f, 0.f, 0.f};
#pragma unroll
        for (int kb = 0; kb < 4; ++kb) {
            Frag16 A, B;
            A.u8 = *(const v8u*)(ysF + (mt * 4 + kb) * 256 + lane * 8);
            B.u8 = *(const v8u*)(w1f + (nt * 4 + kb) * 256 + lane * 8);
            c = __builtin_amdgcn_wmma_f32_16x16x32_bf16(
                    false, A.v, false, B.v, (short)0, c, false, false);
        }
        const int n = nt * 16 + lc;
        const float bias = b1b[n];
        // write h (bf16) into GEMM2 A-frag order
        const int laneA = (((n >> 3) & 1) << 4);
        const int j     = (((n >> 4) & 1) << 2) + ((n & 7) >> 1);
        const int fbase = mt * 8 + (n >> 5);
#pragma unroll
        for (int r = 0; r < 8; ++r) {
            float v = c[r] + bias;
            v = v > 0.f ? v : 0.f;
            const int m = lhi * 8 + r;      // row within 16-tile
            hsF16[fbase * 512 + (laneA + m) * 16 + j * 2 + (n & 1)] = f2bf(v);
        }
    }
    __syncthreads();

    // ---------- Stage 4: GEMM2 (ragged)  out = h[80x256] @ W2[256xN2] + b2 ----------
    const float* b2p = ba.b2[band];
    const int N2  = ba.n2[band];
    const int off = ba.off[band];
    const int n2t = (N2 + 15) >> 4;
    const unsigned int* w2f = wsW2 + ba.w2fo[band];
    float* og = out + ((size_t)bat * TTOT + t0) * OUTC + off;

    for (int tile = wave; tile < MT * n2t; tile += 8) {
        const int mt = tile / n2t;
        const int nt = tile % n2t;
        v8f c = {0.f, 0.f, 0.f, 0.f, 0.f, 0.f, 0.f, 0.f};
#pragma unroll
        for (int kb = 0; kb < 8; ++kb) {
            Frag16 A, B;
            A.u8 = *(const v8u*)(hsF + (mt * 8 + kb) * 256 + lane * 8);
            B.u8 = *(const v8u*)(w2f + (nt * 8 + kb) * 256 + lane * 8);
            c = __builtin_amdgcn_wmma_f32_16x16x32_bf16(
                    false, A.v, false, B.v, (short)0, c, false, false);
        }
        const int col = nt * 16 + lc;
        const bool cv = col < N2;
        const float bias = cv ? b2p[col] : 0.f;
        const int m0 = mt * 16 + lhi * 8;
#pragma unroll
        for (int r = 0; r < 8; ++r) {
            if (cv) og[(size_t)(m0 + r) * OUTC + col] = c[r] + bias;
        }
    }
}

extern "C" void kernel_launch(void* const* d_in, const int* in_sizes, int n_in,
                              void* d_out, int out_size, void* d_ws, size_t ws_size,
                              hipStream_t stream) {
    static const int BW[NB] = {10,10,10,10,10,
                               20,20,20,20,20,20,20,20,20,20,
                               40,40,40,40,40,40,40,40,40,40,
                               60,60,60,60,60};
    BandArgs ba;
    int off = 0;
    int fo  = 0;   // dword offset into packed-W2 region
    for (int i = 0; i < NB; ++i) {
        ba.w2[i]   = (const float*)d_in[5 + i];
        ba.b2[i]   = (const float*)d_in[35 + i];
        ba.off[i]  = off;
        ba.n2[i]   = 2 * BW[i];
        ba.w2fo[i] = fo;
        off += 2 * BW[i];
        int n2t = (2 * BW[i] + 15) / 16;
        fo += n2t * 8 * 256;
    }
    // ws layout: [0, 491520) dwords = packed W1; [491520, 491520+266240) = packed W2
    unsigned int* wsW1 = (unsigned int*)d_ws;
    unsigned int* wsW2 = wsW1 + 491520;

    prepack_w1<<<dim3(1920), dim3(256), 0, stream>>>((const float*)d_in[3], wsW1);
    prepack_w2<<<dim3(64, NB), dim3(256), 0, stream>>>(ba, wsW2);

    mask_estimation_fused<<<dim3(NB * NBATCH * NCHUNK), dim3(256), 0, stream>>>(
        (const float*)d_in[0],   // x
        (const float*)d_in[1],   // ln_gamma
        (const float*)d_in[2],   // ln_beta
        (const float*)d_in[4],   // b1
        wsW1, wsW2,
        (float*)d_out,
        ba);
}
